// ActorCritic_49409303773525
// MI455X (gfx1250) — matrix operand order repaired and checked by reading, not compiled, not run on hardware
//
#include <hip/hip_runtime.h>
#include <hip/hip_bf16.h>

typedef __attribute__((ext_vector_type(16))) _Float16 v16h;
typedef __attribute__((ext_vector_type(8)))  _Float16 v8h;
typedef __attribute__((ext_vector_type(8)))  float    v8f;

#define T_STEPS 100
#define B_SZ    2048
#define D_IN    128
#define H_SZ    256

// f16 workspace layout (element offsets)
#define W1A_OFF 0
#define W1C_OFF 32768          // H*D
#define W2A_OFF 65536          // padded [32][256]
#define W2C_OFF 73728          // padded [16][256]
#define WH_TOTAL 77824

__device__ __forceinline__ v8f wmma_f16(v16h a, v16h b, v8f c) {
  // D = A(16x32 f16) * B(32x16 f16) + C(16x16 f32)
  return __builtin_amdgcn_wmma_f32_16x16x32_f16(
      /*neg_a=*/false, a, /*neg_b=*/false, b,
      /*c_mod=*/(short)0, c, /*reuse_a=*/false, /*reuse_b=*/false);
}

// 16-bit A-matrix 16x32 fragment from a row-major [16][stride] f16 buffer.
// Contiguous layout: lane m = lane&15; halves [0..15] = K kseg..kseg+15,
// kseg = (lane>>4)*16.  K base = s*32.
__device__ __forceinline__ v16h load_a16(const _Float16* base, int lcol,
                                         int lhalf, int s, int stride) {
  const _Float16* p = base + lcol * stride + s * 32 + lhalf * 16;
  v8h lo = *(const v8h*)p;
  v8h hi = *(const v8h*)(p + 8);
  return __builtin_shufflevector(lo, hi, 0, 1, 2, 3, 4, 5, 6, 7,
                                 8, 9, 10, 11, 12, 13, 14, 15);
}

// Weight prep: f32 -> f16, pad W2a to [32][H], W2c to [16][H].
__global__ void prep_weights(const float* __restrict__ W1a,
                             const float* __restrict__ W1c,
                             const float* __restrict__ W2a,
                             const float* __restrict__ W2c,
                             _Float16* __restrict__ wh) {
  int i = blockIdx.x * blockDim.x + threadIdx.x;
  if (i >= WH_TOTAL) return;
  if (i < W1C_OFF) {
    wh[i] = (_Float16)W1a[i];
  } else if (i < W2A_OFF) {
    wh[i] = (_Float16)W1c[i - W1C_OFF];
  } else if (i < W2C_OFF) {
    int k = i - W2A_OFF;
    int j = k >> 8;  // row in padded [32][256]
    wh[i] = (_Float16)((j < 20) ? W2a[k] : 0.0f);
  } else {
    int k = i - W2C_OFF;
    int j = k >> 8;  // row in padded [16][256]
    wh[i] = (_Float16)((j < 1) ? W2c[k] : 0.0f);
  }
}

__global__ __launch_bounds__(512) void snn_fused(
    const float* __restrict__ spikes,  // [T,B,D] f32 (0/1)
    const float* __restrict__ b1a, const float* __restrict__ b2a,
    const float* __restrict__ b1c, const float* __restrict__ b2c,
    const _Float16* __restrict__ wh, float* __restrict__ out) {
  const int tid   = (int)threadIdx.x;
  const int wave  = tid >> 5;    // 16 waves (wave32)
  const int lane  = tid & 31;
  const int lhalf = lane >> 4;   // 0/1
  const int lcol  = lane & 15;
  const int b0    = (int)blockIdx.x * 16;

  __shared__ __align__(16) _Float16 xs[16 * 128];        // x_t tile (f16)
  __shared__ __align__(16) _Float16 spkS[2 * 16 * 256];  // layer1 spikes a/c
  __shared__ __align__(16) float    red[16 * 32];        // acc reduction

  const _Float16* w1a = wh + W1A_OFF;
  const _Float16* w1c = wh + W1C_OFF;
  const _Float16* w2a = wh + W2A_OFF;
  const _Float16* w2c = wh + W2C_OFF;

  // Layer1: wave w owns output columns h = 16w .. 16w+15 for both nets.
  // B-fragment (32x16, 16-bit): per lane column n = lane&15 -> weight row
  // h = 16w + n; contiguous 16 halves at K offset s*32 + (lane>>4)*16.
  const int hrow = wave * 16 + lcol;
  v16h wAf[4], wCf[4];
#pragma unroll
  for (int s = 0; s < 4; ++s) {
    wAf[s] = *(const v16h*)(w1a + hrow * 128 + s * 32 + lhalf * 16);
    wCf[s] = *(const v16h*)(w1c + hrow * 128 + s * 32 + lhalf * 16);
  }
  const float ba1 = b1a[hrow];
  const float bc1 = b1c[hrow];

  // Membrane state lives in WMMA C/D layout: element r of v8f is row
  // m = r + 8*(lane>>4), column n = lane&15 (fixed across t).
  v8f m_a1 = {0, 0, 0, 0, 0, 0, 0, 0};
  v8f m_c1 = {0, 0, 0, 0, 0, 0, 0, 0};
  v8f m2   = {0, 0, 0, 0, 0, 0, 0, 0};
  v8f acc2 = {0, 0, 0, 0, 0, 0, 0, 0};

  // Layer2 role assignment (wave-uniform so EXEC stays all-ones at WMMA):
  //  wave0: actor cols 0..15, wave1: actor cols 16..31 (>=20 padded),
  //  wave2: critic cols 0..15 (>=1 padded).
  const _Float16* w2base = nullptr;
  int   jrow  = 0;
  float bias2 = 0.0f;
  if (wave == 0) { jrow = lcol;      w2base = w2a; bias2 = b2a[jrow]; }
  if (wave == 1) { jrow = 16 + lcol; w2base = w2a; bias2 = (jrow < 20) ? b2a[jrow] : 0.0f; }
  if (wave == 2) { jrow = lcol;      w2base = w2c; bias2 = (lcol == 0) ? b2c[0] : 0.0f; }

  for (int t = 0; t < T_STEPS; ++t) {
    __syncthreads();  // prev layer2 done with spkS; xs consumers done
    {
      // Stage x_t[16 rows][128] f32 -> f16 in LDS. 512 thr x 4 elems.
      int base = tid * 4;
      int m = base >> 7;
      int d = base & 127;
      const float4 v = *(const float4*)(spikes + (size_t)t * B_SZ * D_IN +
                                        (size_t)(b0 + m) * D_IN + d);
      xs[m * 128 + d + 0] = (_Float16)v.x;
      xs[m * 128 + d + 1] = (_Float16)v.y;
      xs[m * 128 + d + 2] = (_Float16)v.z;
      xs[m * 128 + d + 3] = (_Float16)v.w;
      if (t + 1 < T_STEPS)  // hide next step's HBM latency
        __builtin_prefetch(spikes + (size_t)(t + 1) * B_SZ * D_IN +
                               (size_t)(b0 + m) * D_IN + d, 0, 0);
    }
    __syncthreads();

    // ---- Layer 1: [16x128] @ [128x16] per wave, actor + critic ----
    v8f za = {0, 0, 0, 0, 0, 0, 0, 0};
    v8f zc = {0, 0, 0, 0, 0, 0, 0, 0};
#pragma unroll
    for (int s = 0; s < 4; ++s) {
      v16h a = load_a16(xs, lcol, lhalf, s, 128);
      za = wmma_f16(a, wAf[s], za);
      zc = wmma_f16(a, wCf[s], zc);
    }
    // Leaky step (reset-by-subtract from previous membrane), emit spikes.
#pragma unroll
    for (int r = 0; r < 8; ++r) {
      int mrow = r + 8 * lhalf;
      float cur = za[r] + ba1;
      float rst = (m_a1[r] > 1.0f) ? 1.0f : 0.0f;
      float mn  = 0.95f * m_a1[r] + cur - rst;
      m_a1[r]   = mn;
      spkS[mrow * 256 + hrow] = (_Float16)(((mn - 1.0f) > 0.0f) ? 1.0f : 0.0f);

      cur = zc[r] + bc1;
      rst = (m_c1[r] > 1.0f) ? 1.0f : 0.0f;
      mn  = 0.95f * m_c1[r] + cur - rst;
      m_c1[r] = mn;
      spkS[4096 + mrow * 256 + hrow] =
          (_Float16)(((mn - 1.0f) > 0.0f) ? 1.0f : 0.0f);
    }
    __syncthreads();

    // ---- Layer 2: [16x256] @ [256x16] on waves 0..2 ----
    if (wave < 3) {
      const _Float16* sb = spkS + ((wave == 2) ? 4096 : 0);
      v8f z2 = {0, 0, 0, 0, 0, 0, 0, 0};
#pragma unroll
      for (int s = 0; s < 8; ++s) {
        v16h a = load_a16(sb, lcol, lhalf, s, 256);
        v16h b = *(const v16h*)(w2base + jrow * 256 + s * 32 + lhalf * 16);
        z2 = wmma_f16(a, b, z2);
      }
#pragma unroll
      for (int r = 0; r < 8; ++r) {
        float cur = z2[r] + bias2;
        float rst = (m2[r] > 1.0f) ? 1.0f : 0.0f;
        float mn  = 0.95f * m2[r] + cur - rst;
        m2[r]     = mn;
        float spk = ((mn - 1.0f) > 0.0f) ? 1.0f : 0.0f;
        if (wave < 2) acc2[r] += spk;  // population spike accumulator
      }
    }
  }

  __syncthreads();
  // Spill actor accumulators to LDS for the population reduction.
  if (wave < 2) {
#pragma unroll
    for (int r = 0; r < 8; ++r)
      red[(r + 8 * lhalf) * 32 + wave * 16 + lcol] = acc2[r];
  }
  // Critic membrane out: m_c2 is column 0 of wave2's tile.
  if (wave == 2 && lcol == 0) {
#pragma unroll
    for (int r = 0; r < 8; ++r)
      out[2 * B_SZ + b0 + r + 8 * lhalf] = m2[r];
  }
  __syncthreads();

  // Population-coded softmax over 2 actions, 16 rows per block.
  if (tid < 16) {
    float a1 = 0.0f, a2 = 0.0f;
#pragma unroll
    for (int j = 0; j < 10; ++j)  a1 += red[tid * 32 + j];
#pragma unroll
    for (int j = 10; j < 20; ++j) a2 += red[tid * 32 + j];
    float mx = fmaxf(a1, a2);
    float e1 = __expf(a1 - mx);
    float e2 = __expf(a2 - mx);
    float inv = 1.0f / (e1 + e2);
    out[(b0 + tid) * 2 + 0] = e1 * inv;
    out[(b0 + tid) * 2 + 1] = e2 * inv;
  }
}

extern "C" void kernel_launch(void* const* d_in, const int* in_sizes, int n_in,
                              void* d_out, int out_size, void* d_ws,
                              size_t ws_size, hipStream_t stream) {
  (void)in_sizes; (void)n_in; (void)out_size; (void)ws_size;
  const float* spikes = (const float*)d_in[0];
  const float* W1a    = (const float*)d_in[1];
  const float* b1a    = (const float*)d_in[2];
  const float* W2a    = (const float*)d_in[3];
  const float* b2a    = (const float*)d_in[4];
  const float* W1c    = (const float*)d_in[5];
  const float* b1c    = (const float*)d_in[6];
  const float* W2c    = (const float*)d_in[7];
  const float* b2c    = (const float*)d_in[8];
  float*     out = (float*)d_out;
  _Float16*  wh  = (_Float16*)d_ws;  // needs 77824*2 = 155,648 bytes

  prep_weights<<<(WH_TOTAL + 255) / 256, 256, 0, stream>>>(W1a, W1c, W2a, W2c, wh);
  snn_fused<<<B_SZ / 16, 512, 0, stream>>>(spikes, b1a, b2a, b1c, b2c, wh, out);
}
